// SelfAttention_51196010168887
// MI455X (gfx1250) — matrix operand order repaired and checked by reading, not compiled, not run on hardware
//
#include <hip/hip_runtime.h>
#include <hip/hip_bf16.h>
#include <stdint.h>

// ---------------------------------------------------------------------------
// Self-attention + residual + LayerNorm for MI455X (gfx1250, wave32, WMMA).
// B=4, N=4096, D=512, fp32 in/out; internally f16 WMMA with f32 accumulate.
// Flash attention double-buffers K blocks in LDS via async global->LDS
// copies (ASYNCcnt) overlapped with WMMA compute.
// ---------------------------------------------------------------------------

typedef _Float16 v16h __attribute__((ext_vector_type(16)));
typedef float    v8f  __attribute__((ext_vector_type(8)));

#define BATCH 4
#define SEQ   4096
#define DIM   512
#define ROWS  (BATCH * SEQ)      // 16384
#define LN_EPS 1e-5f
// softmax in base-2: scale = 1/sqrt(512) * log2(e)
#define SCALE_LOG2E 0.063758351f

// ---- A-operand fragment (16x32 f16, M x K), per CDNA5 ISA layout ----------
// lane<16:  row=lane,    K = {0..7} in v0..3, {16..23} in v4..7
// lane>=16: row=lane-16, K = {8..15},{24..31}
__device__ inline v16h frag_ld_a(const _Float16* base, int rowStride, int lane) {
  const int r  = lane & 15;
  const int kb = (lane & 16) ? 8 : 0;
  const _Float16* p = base + (size_t)r * rowStride + kb;
  v16h f;
#pragma unroll
  for (int v = 0; v < 8; ++v) {
    const int k = ((v & 4) ? 16 : 0) + 2 * (v & 3);
    f[2 * v]     = p[k];
    f[2 * v + 1] = p[k + 1];
  }
  return f;
}

// ---- B-operand fragment (32x16 f16, K x N), stored as B^T tile [n][k] -----
// lane<16:  col n=lane,    K = {0..7},{16..23}
// lane>=16: col n=lane-16, K = {8..15},{24..31}
__device__ inline v16h frag_ld_b(const _Float16* base, int rowStride, int lane) {
  const int n  = lane & 15;
  const int kb = (lane & 16) ? 8 : 0;
  const _Float16* p = base + (size_t)n * rowStride + kb;
  v16h f;
#pragma unroll
  for (int v = 0; v < 16; ++v) f[v] = p[v < 8 ? v : v + 8];
  return f;
}

__device__ inline v8f wmma_f16(v16h a, v16h b, v8f c) {
  return __builtin_amdgcn_wmma_f32_16x16x32_f16(false, a, false, b, (short)0, c,
                                                false, false);
}

// Async global -> LDS copy of 16 bytes (GLOBAL_LOAD_ASYNC_TO_LDS_B128, GV
// mode, tracked by ASYNCcnt). lds_off = byte offset within workgroup LDS.
__device__ inline void async_copy_b128(uint32_t lds_off, uint64_t gaddr) {
  asm volatile("global_load_async_to_lds_b128 %0, %1, off"
               :: "v"(lds_off), "v"(gaddr)
               : "memory");
}
__device__ inline void wait_asynccnt0() {
  asm volatile("s_wait_asynccnt 0x0" ::: "memory");
}

// XOR-lane exchange via DS_SWIZZLE_B32 (group-of-32 mode: and=0x1f, xor=MASK).
template <int MASK>
__device__ inline float swz_xor(float x) {
  return __int_as_float(
      __builtin_amdgcn_ds_swizzle(__float_as_int(x), (MASK << 10) | 0x1f));
}

// ---------------------------------------------------------------------------
// Kernel 1: weight transpose + cast:  WT[e][d] = (f16) W[d][e]
// ---------------------------------------------------------------------------
__global__ void wtrans_kernel(const float* __restrict__ W,
                              _Float16* __restrict__ WT) {
  const int d = blockIdx.x * blockDim.x + threadIdx.x;  // 0..511
  const int e = blockIdx.y;                             // 0..511
  WT[(size_t)e * DIM + d] = (_Float16)W[(size_t)d * DIM + e];
}

// ---------------------------------------------------------------------------
// Kernel 2: out = x @ W  (via WT), 64x64 tile per block, 4 waves.
// x is fp32 [ROWS][DIM]; WT is f16 [DIM][DIM] (n-major, k contiguous).
// TRANS=0: out f16 [ROWS][DIM];  TRANS=1: out f16 [B][DIM][SEQ] (Vt).
// ---------------------------------------------------------------------------
template <int TRANS>
__global__ __launch_bounds__(128) void gemm_xw_kernel(
    const float* __restrict__ X, const _Float16* __restrict__ WT,
    _Float16* __restrict__ out) {
  __shared__ __align__(16) _Float16 Al[64 * 32];   // A tile   [m][k]
  __shared__ __align__(16) _Float16 Bl[64 * 32];   // B^T tile [n][k]
  const int tid  = threadIdx.x;
  const int lane = tid & 31;
  const int w    = tid >> 5;
  const int rowBase = blockIdx.x * 64;
  const int colBase = blockIdx.y * 64;

  v8f acc[4] = {};
  uint32_t* Bl32 = (uint32_t*)Bl;
  const uint32_t* WT32 = (const uint32_t*)WT;

  for (int k0 = 0; k0 < DIM; k0 += 32) {
    __syncthreads();
    // stage A: 64x32 halves, converting fp32 -> f16 on the fly
#pragma unroll
    for (int i = 0; i < 16; ++i) {
      const int idx = tid * 16 + i;             // 0..2047
      const int r = idx >> 5, c = idx & 31;
      Al[idx] = (_Float16)X[(size_t)(rowBase + r) * DIM + k0 + c];
    }
    // stage B^T: 64x32 halves = 1024 dwords
#pragma unroll
    for (int i = 0; i < 8; ++i) {
      const int idx = tid * 8 + i;              // 0..1023
      const int r = idx >> 4, c = idx & 15;
      Bl32[idx] = WT32[(size_t)(colBase + r) * (DIM / 2) + (k0 >> 1) + c];
    }
    __syncthreads();

    const v16h af = frag_ld_a(Al + w * 16 * 32, 32, lane);
#pragma unroll
    for (int t = 0; t < 4; ++t) {
      const v16h bf = frag_ld_b(Bl + t * 16 * 32, 32, lane);
      acc[t] = wmma_f16(af, bf, acc[t]);
    }
  }

  // store (C layout: vgpr r -> row (lane<16 ? r : r+8), col = lane&15)
  const int mr = (lane & 16) ? 8 : 0;
  const int nc = lane & 15;
#pragma unroll
  for (int t = 0; t < 4; ++t) {
#pragma unroll
    for (int r = 0; r < 8; ++r) {
      const int gr = rowBase + w * 16 + mr + r;
      const int gc = colBase + t * 16 + nc;
      const _Float16 val = (_Float16)acc[t][r];
      if (TRANS == 0) {
        out[(size_t)gr * DIM + gc] = val;
      } else {
        const int b = gr >> 12, n = gr & 4095;
        out[((size_t)b * DIM + gc) * SEQ + n] = val;
      }
    }
  }
}

// ---------------------------------------------------------------------------
// Kernel 3: flash attention.  block = 128 threads (4 waves), 64 query rows.
// Each wave: 16 query rows, full D=512, online softmax over 32-key blocks.
// Q fragments cached in registers (gfx1250 >256-VGPR budget).
// K blocks double-buffered in dynamic LDS; async global->LDS copies for
// block j+1 overlap the WMMA compute of block j (waited via ASYNCcnt).
// Qh,Kh: f16 [B][N][D];  Vt: f16 [B][D][N];  att: f32 [B][N][D].
// ---------------------------------------------------------------------------
#define KBUF (32 * DIM)                 // halves per K buffer (32 KB)
#define FLASH_LDS_BYTES ((2 * KBUF + 4 * 16 * 32) * 2)  // 69632 B

__global__ __launch_bounds__(128) void flash_attn_kernel(
    const _Float16* __restrict__ Qh, const _Float16* __restrict__ Kh,
    const _Float16* __restrict__ Vt, float* __restrict__ att) {
  extern __shared__ __align__(16) _Float16 smem[];
  _Float16* Kbuf[2] = {smem, smem + KBUF};
  _Float16* Pl = smem + 2 * KBUF;       // per-wave P tiles (4 KB)

  const int tid  = threadIdx.x;
  const int lane = tid & 31;
  const int w    = tid >> 5;
  const int b     = blockIdx.y;
  const int qBase = blockIdx.x * 64;

  // ---- cache the wave's 16 Q rows as 16 A-fragments (128 VGPRs) ----
  const _Float16* qrow = Qh + ((size_t)b * SEQ + qBase + w * 16) * DIM;
  v16h qf[16];
#pragma unroll
  for (int kk = 0; kk < 16; ++kk) qf[kk] = frag_ld_a(qrow + kk * 32, DIM, lane);

  v8f o[32] = {};
  float m[8], l[8];
#pragma unroll
  for (int r = 0; r < 8; ++r) { m[r] = -3.0e38f; l[r] = 0.0f; }

  const uint32_t klds0 = (uint32_t)(uintptr_t)Kbuf[0];
  const uint32_t klds1 = (uint32_t)(uintptr_t)Kbuf[1];
  const uint64_t kgbase = (uint64_t)(uintptr_t)(Kh + (size_t)b * SEQ * DIM);

  // helper: issue 16 async b128 copies of one 32-row K block (32 KB)
  auto stage_k = [&](uint32_t ldsb, int kb) {
    const uint64_t g = kgbase + (uint64_t)kb * (DIM * 2);
#pragma unroll
    for (int i = 0; i < 16; ++i) {
      const uint32_t off = (uint32_t)(tid + i * 128) * 16u;
      async_copy_b128(ldsb + off, g + off);
    }
  };

  // prologue: stage block 0
  stage_k(klds0, 0);
  wait_asynccnt0();
  __syncthreads();

  for (int j = 0; j < SEQ / 32; ++j) {
    const int kb = j * 32;
    const _Float16* Kl = (j & 1) ? Kbuf[1] : Kbuf[0];

    // ---- kick off async stage of block j+1 into the other buffer ----
    if (j + 1 < SEQ / 32) {
      stage_k((j & 1) ? klds0 : klds1, kb + 32);
      // warm L2 two blocks ahead
      if (j + 2 < SEQ / 32) {
        const char* nk = (const char*)(Kh + ((size_t)b * SEQ + kb + 64) * DIM);
        __builtin_prefetch(nk + tid * 256, 0, 1);
      }
    }

    // ---- S = Q K^T over full D (16 k-steps), two 16-key tiles ----
    v8f s0 = {}, s1 = {};
#pragma unroll
    for (int kk = 0; kk < 16; ++kk) {
      const v16h b0 = frag_ld_b(Kl + 0 * 16 * DIM + kk * 32, DIM, lane);
      const v16h b1 = frag_ld_b(Kl + 16 * DIM + kk * 32, DIM, lane);
      s0 = wmma_f16(qf[kk], b0, s0);
      s1 = wmma_f16(qf[kk], b1, s1);
    }

    // ---- online softmax (base-2), half-wave XOR reductions (ds_swizzle) ----
    float corr[8];
#pragma unroll
    for (int r = 0; r < 8; ++r) {
      float a0 = s0[r] * SCALE_LOG2E;
      float a1 = s1[r] * SCALE_LOG2E;
      float mx = fmaxf(a0, a1);
      mx = fmaxf(mx, swz_xor<1>(mx));
      mx = fmaxf(mx, swz_xor<2>(mx));
      mx = fmaxf(mx, swz_xor<4>(mx));
      mx = fmaxf(mx, swz_xor<8>(mx));
      const float mn = fmaxf(m[r], mx);
      const float c  = __builtin_exp2f(m[r] - mn);
      m[r] = mn;
      const float p0 = __builtin_exp2f(a0 - mn);
      const float p1 = __builtin_exp2f(a1 - mn);
      s0[r] = p0; s1[r] = p1;
      float rs = p0 + p1;
      rs += swz_xor<1>(rs);
      rs += swz_xor<2>(rs);
      rs += swz_xor<4>(rs);
      rs += swz_xor<8>(rs);
      l[r] = l[r] * c + rs;
      corr[r] = c;
    }
#pragma unroll
    for (int dt = 0; dt < 32; ++dt)
#pragma unroll
      for (int r = 0; r < 8; ++r) o[dt][r] *= corr[r];

    // ---- P: C-layout regs -> LDS -> A-layout fragment ----
    {
      _Float16* Pw = Pl + w * 16 * 32;
      const int mr = (lane & 16) ? 8 : 0;
      const int nc = lane & 15;
#pragma unroll
      for (int r = 0; r < 8; ++r) {
        Pw[(mr + r) * 32 + nc]      = (_Float16)s0[r];
        Pw[(mr + r) * 32 + 16 + nc] = (_Float16)s1[r];
      }
    }
    const v16h pf = frag_ld_a(Pl + w * 16 * 32, 32, lane);

    // ---- O += P V : B fragments straight from Vt (contiguous in n) ----
    const _Float16* vbase = Vt + (size_t)b * DIM * SEQ + kb;
#pragma unroll
    for (int dt = 0; dt < 32; ++dt) {
      const v16h vf = frag_ld_b(vbase + (size_t)(dt * 16) * SEQ, SEQ, lane);
      o[dt] = wmma_f16(pf, vf, o[dt]);
    }

    // ---- retire this iteration's async copies, publish next buffer ----
    wait_asynccnt0();
    __syncthreads();
  }

  // ---- normalize and store f32 ----
  const int mr = (lane & 16) ? 8 : 0;
  const int nc = lane & 15;
  float* outp = att + ((size_t)b * SEQ + qBase + w * 16) * DIM;
#pragma unroll
  for (int r = 0; r < 8; ++r) {
    const float inv = 1.0f / l[r];
#pragma unroll
    for (int dt = 0; dt < 32; ++dt)
      outp[(size_t)(mr + r) * DIM + dt * 16 + nc] = o[dt][r] * inv;
  }
}

// ---------------------------------------------------------------------------
// Kernel 4: out = LayerNorm(att + x) * gamma + beta.  One wave per row.
// ---------------------------------------------------------------------------
__global__ __launch_bounds__(256) void resid_ln_kernel(
    const float* __restrict__ att, const float* __restrict__ x,
    const float* __restrict__ gamma, const float* __restrict__ beta,
    float* __restrict__ out) {
  const int lane = threadIdx.x & 31;
  const int wv   = threadIdx.x >> 5;
  const int row  = blockIdx.x * 8 + wv;
  const size_t base = (size_t)row * DIM;

  float h[16];
  float s = 0.0f, ss = 0.0f;
#pragma unroll
  for (int i = 0; i < 16; ++i) {
    const int c = lane + i * 32;
    const float v = att[base + c] + x[base + c];
    h[i] = v; s += v; ss += v * v;
  }
#pragma unroll
  for (int msk = 1; msk < 32; msk <<= 1) {
    s  += __shfl_xor(s, msk, 32);
    ss += __shfl_xor(ss, msk, 32);
  }
  const float mu   = s * (1.0f / DIM);
  const float var  = ss * (1.0f / DIM) - mu * mu;
  const float rstd = rsqrtf(var + LN_EPS);
#pragma unroll
  for (int i = 0; i < 16; ++i) {
    const int c = lane + i * 32;
    out[base + c] = (h[i] - mu) * rstd * gamma[c] + beta[c];
  }
}

// ---------------------------------------------------------------------------
// Launch
// ---------------------------------------------------------------------------
extern "C" void kernel_launch(void* const* d_in, const int* in_sizes, int n_in,
                              void* d_out, int out_size, void* d_ws,
                              size_t ws_size, hipStream_t stream) {
  const float* x     = (const float*)d_in[0];
  const float* Wq    = (const float*)d_in[1];
  const float* Wk    = (const float*)d_in[2];
  const float* Wv    = (const float*)d_in[3];
  const float* gamma = (const float*)d_in[4];
  const float* beta  = (const float*)d_in[5];
  float* out = (float*)d_out;

  // workspace carve-out (~86 MB)
  char* ws = (char*)d_ws;
  _Float16* WqT = (_Float16*)ws; ws += (size_t)DIM * DIM * 2;
  _Float16* WkT = (_Float16*)ws; ws += (size_t)DIM * DIM * 2;
  _Float16* WvT = (_Float16*)ws; ws += (size_t)DIM * DIM * 2;
  _Float16* Qh  = (_Float16*)ws; ws += (size_t)ROWS * DIM * 2;
  _Float16* Kh  = (_Float16*)ws; ws += (size_t)ROWS * DIM * 2;
  _Float16* Vt  = (_Float16*)ws; ws += (size_t)ROWS * DIM * 2;
  float*    att = (float*)ws;    ws += (size_t)ROWS * DIM * 4;

  // 1) weight transpose + cast
  {
    dim3 g(DIM / 256, DIM);
    wtrans_kernel<<<g, 256, 0, stream>>>(Wq, WqT);
    wtrans_kernel<<<g, 256, 0, stream>>>(Wk, WkT);
    wtrans_kernel<<<g, 256, 0, stream>>>(Wv, WvT);
  }
  // 2) projections (V written transposed)
  {
    dim3 g(ROWS / 64, DIM / 64);
    gemm_xw_kernel<0><<<g, 128, 0, stream>>>(x, WqT, Qh);
    gemm_xw_kernel<0><<<g, 128, 0, stream>>>(x, WkT, Kh);
    gemm_xw_kernel<1><<<g, 128, 0, stream>>>(x, WvT, Vt);
  }
  // 3) flash attention (double-buffered K staging in dynamic LDS)
  {
    dim3 g(SEQ / 64, BATCH);
    flash_attn_kernel<<<g, 128, FLASH_LDS_BYTES, stream>>>(Qh, Kh, Vt, att);
  }
  // 4) residual + LayerNorm
  resid_ln_kernel<<<ROWS / 8, 256, 0, stream>>>(att, x, gamma, beta, out);
}
